// SparseAttention_88484916232472
// MI455X (gfx1250) — compile-verified, matrix operand order checked by reading, not run on hardware
//
#include <hip/hip_runtime.h>
#include <hip/hip_bf16.h>

// Problem constants (from the reference)
constexpr int BATCH = 2;
constexpr int LSEQ  = 2048;
constexpr int DMODEL= 1024;
constexpr int NHEAD = 16;
constexpr int HDIM  = 64;        // DMODEL / NHEAD
constexpr int HALFW = 128;       // WINDOW/2, |i-j| <= HALFW allowed
constexpr int N3D   = 3 * DMODEL;

typedef _Float16 v16h __attribute__((ext_vector_type(16)));
typedef _Float16 v8h  __attribute__((ext_vector_type(8)));
typedef float    v8f  __attribute__((ext_vector_type(8)));

union ABfrag { v16h v; v8h h[2]; };

static __device__ __forceinline__ v8f wmma_f16(const ABfrag& a, const ABfrag& b, v8f c) {
    // (neg_a, A, neg_b, B, c_mod, C, reuse_a, reuse_b)
    return __builtin_amdgcn_wmma_f32_16x16x32_f16(false, a.v, false, b.v, (short)0, c,
                                                  false, false);
}

static __device__ __forceinline__ void load_frag(ABfrag& f, const _Float16* p) {
    f.h[0] = *(const v8h*)(p);
    f.h[1] = *(const v8h*)(p + 16);
}

// ---------------------------------------------------------------------------
// Prep: fp32 -> f16 elementwise
// ---------------------------------------------------------------------------
__global__ void cvt_f32_f16(const float* __restrict__ in, _Float16* __restrict__ out, int n) {
    int i = blockIdx.x * blockDim.x + threadIdx.x;
    if (i < n) out[i] = (_Float16)in[i];
}

// Prep: transpose [K,N] fp32 -> [N,K] f16 (so WMMA B-fragments load contiguously)
__global__ void transpose_f32_f16(const float* __restrict__ in, _Float16* __restrict__ out,
                                  int K, int N) {
    int i = blockIdx.x * blockDim.x + threadIdx.x;
    if (i < K * N) {
        int n = i / K;
        int k = i - n * K;
        out[i] = (_Float16)in[(size_t)k * N + n];
    }
}

// ---------------------------------------------------------------------------
// Shared GEMM main loop: one wave computes a 32(M) x 64(N) tile of A @ B^T'
// (BT is [N,K] row-major).  Software-pipelined: next K-chunk's fragments are
// loaded before the current chunk's 8 WMMAs, hiding load latency behind the
// matrix pipe.  acc[i][t]: i = M half (0:+0..15, 1:+16..31), t = N quarter.
// ---------------------------------------------------------------------------
static __device__ __forceinline__ void gemm_tile_32x64(
        const _Float16* __restrict__ A, const _Float16* __restrict__ BT,
        int K, int m0, int n0, int r, int koff, v8f acc[2][4]) {
    const _Float16* arow0 = A + (size_t)(m0 + r) * K + koff;
    const _Float16* arow1 = A + (size_t)(m0 + 16 + r) * K + koff;
    const _Float16* brow0 = BT + (size_t)(n0 +  0 + r) * K + koff;
    const _Float16* brow1 = BT + (size_t)(n0 + 16 + r) * K + koff;
    const _Float16* brow2 = BT + (size_t)(n0 + 32 + r) * K + koff;
    const _Float16* brow3 = BT + (size_t)(n0 + 48 + r) * K + koff;

    ABfrag ac[2], bc[4];
    load_frag(ac[0], arow0);
    load_frag(ac[1], arow1);
    load_frag(bc[0], brow0);
    load_frag(bc[1], brow1);
    load_frag(bc[2], brow2);
    load_frag(bc[3], brow3);

    for (int k0 = 0; k0 < K; k0 += 32) {
        const int kn = (k0 + 32 < K) ? (k0 + 32) : 0;   // dummy reload on last iter
        ABfrag an[2], bn[4];
        load_frag(an[0], arow0 + kn);
        load_frag(an[1], arow1 + kn);
        load_frag(bn[0], brow0 + kn);
        load_frag(bn[1], brow1 + kn);
        load_frag(bn[2], brow2 + kn);
        load_frag(bn[3], brow3 + kn);
        __builtin_prefetch(arow0 + kn + 128, 0, 1);

#pragma unroll
        for (int i = 0; i < 2; ++i)
#pragma unroll
            for (int t = 0; t < 4; ++t)
                acc[i][t] = wmma_f16(ac[i], bc[t], acc[i][t]);

        ac[0] = an[0]; ac[1] = an[1];
        bc[0] = bn[0]; bc[1] = bn[1]; bc[2] = bn[2]; bc[3] = bn[3];
    }
}

// ---------------------------------------------------------------------------
// QKV projection:  qkv[m,n] = sum_k x[m,k] * w_qkv[k,n] + b[n]
// One wave computes a 32(M) x 64(N) tile.  Outputs scattered into
//   q : [B,H,L,HD]   k : [B,H,L,HD]   vT : [B,H,HD,L]
// ---------------------------------------------------------------------------
__global__ __launch_bounds__(32)
void qkv_gemm(const _Float16* __restrict__ xh, const _Float16* __restrict__ wT,
              const float* __restrict__ bias,
              _Float16* __restrict__ qo, _Float16* __restrict__ ko, _Float16* __restrict__ vo) {
    const int lane = threadIdx.x;
    const int r    = lane & 15;
    const int hf   = lane >> 4;
    const int koff = hf * 8;
    const int m0   = blockIdx.y * 32;
    const int n0   = blockIdx.x * 64;

    v8f acc[2][4];
#pragma unroll
    for (int i = 0; i < 2; ++i)
#pragma unroll
        for (int t = 0; t < 4; ++t)
#pragma unroll
            for (int g = 0; g < 8; ++g) acc[i][t][g] = 0.0f;

    gemm_tile_32x64(xh, wT, DMODEL, m0, n0, r, koff, acc);

#pragma unroll
    for (int t = 0; t < 4; ++t) {
        const int n     = n0 + t * 16 + r;
        const float bb  = bias[n];
        const int which = n >> 10;         // 0=q 1=k 2=v
        const int d     = n & (DMODEL - 1);
        const int h     = d >> 6;
        const int hd    = d & (HDIM - 1);
        const size_t bhbase = (size_t)h;   // combined below with batch
#pragma unroll
        for (int i = 0; i < 2; ++i) {
            const int mbase = m0 + i * 16 + hf * 8;
#pragma unroll
            for (int g = 0; g < 8; ++g) {
                const int m = mbase + g;
                const int b = m >> 11;                 // / LSEQ
                const int l = m & (LSEQ - 1);
                const _Float16 val = (_Float16)(acc[i][t][g] + bb);
                const size_t bh = (size_t)b * NHEAD + bhbase;
                if (which == 0)      qo[(bh * LSEQ + l) * HDIM + hd] = val;
                else if (which == 1) ko[(bh * LSEQ + l) * HDIM + hd] = val;
                else                 vo[(bh * HDIM + hd) * LSEQ + l] = val;
            }
        }
    }
}

// ---------------------------------------------------------------------------
// Banded flash attention.  One wave per (b, h, 16-query tile).
// Processes 32 keys per iteration: QK^T via 2x2 WMMA, online softmax,
// P transposed to A-layout through LDS, PV via 4 WMMA into fp32 acc.
// ctx written as f16 in [B,L,D] layout for the output GEMM.
// ---------------------------------------------------------------------------
__global__ __launch_bounds__(32)
void attn_window(const _Float16* __restrict__ qi, const _Float16* __restrict__ ki,
                 const _Float16* __restrict__ vTi, _Float16* __restrict__ ctx) {
    __shared__ __align__(16) _Float16 pbuf[16 * 32];

    const int lane = threadIdx.x;
    const int r    = lane & 15;
    const int hf   = lane >> 4;
    const int koff = hf * 8;
    const int q0   = blockIdx.x * 16;
    const int h    = blockIdx.y;
    const int b    = blockIdx.z;
    const size_t bh = (size_t)b * NHEAD + h;

    const _Float16* qbase = qi  + bh * LSEQ * HDIM;
    const _Float16* kbase = ki  + bh * LSEQ * HDIM;
    const _Float16* vbase = vTi + bh * HDIM * LSEQ;

    // Persistent Q A-fragments (two K=32 chunks cover HD=64)
    ABfrag qa[2];
    {
        const _Float16* qrow = qbase + (size_t)(q0 + r) * HDIM;
        load_frag(qa[0], qrow + koff);
        load_frag(qa[1], qrow + 32 + koff);
    }

    v8f acc[4];
#pragma unroll
    for (int t = 0; t < 4; ++t)
#pragma unroll
        for (int g = 0; g < 8; ++g) acc[t][g] = 0.0f;
    float rm[8], rl[8];
#pragma unroll
    for (int g = 0; g < 8; ++g) { rm[g] = -1.0e30f; rl[g] = 0.0f; }

    int js = q0 - HALFW; if (js < 0) js = 0; js &= ~31;          // 32-aligned, in range
    int je = q0 + 16 + HALFW; if (je > LSEQ) je = LSEQ;

    const float scale = 0.125f;   // 1/sqrt(64)

    for (int j0 = js; j0 < je; j0 += 32) {
        // ---- scores for keys [j0, j0+32) ----
        v8f s[2];
#pragma unroll
        for (int c = 0; c < 2; ++c) {
#pragma unroll
            for (int g = 0; g < 8; ++g) s[c][g] = 0.0f;
            const _Float16* krow = kbase + (size_t)(j0 + c * 16 + r) * HDIM;
            ABfrag kb0, kb1;
            load_frag(kb0, krow + koff);
            load_frag(kb1, krow + 32 + koff);
            s[c] = wmma_f16(qa[0], kb0, s[c]);
            s[c] = wmma_f16(qa[1], kb1, s[c]);
        }

        // ---- scale + band mask ----
#pragma unroll
        for (int g = 0; g < 8; ++g) {
            const int i = q0 + hf * 8 + g;
#pragma unroll
            for (int c = 0; c < 2; ++c) {
                const int j = j0 + c * 16 + r;
                float sv = s[c][g] * scale;
                int dd = i - j; if (dd < 0) dd = -dd;
                if (dd > HALFW) sv = -1.0e30f;
                s[c][g] = sv;
            }
        }

        // ---- online softmax (row = VGPR slot within lane-half) ----
#pragma unroll
        for (int g = 0; g < 8; ++g) {
            float tm = fmaxf(s[0][g], s[1][g]);
#pragma unroll
            for (int off = 8; off >= 1; off >>= 1)
                tm = fmaxf(tm, __shfl_xor(tm, off, 16));
            const float nm   = fmaxf(rm[g], tm);
            const float corr = __expf(rm[g] - nm);
            const float p0 = (s[0][g] > -1.0e29f) ? __expf(s[0][g] - nm) : 0.0f;
            const float p1 = (s[1][g] > -1.0e29f) ? __expf(s[1][g] - nm) : 0.0f;
            float rs = p0 + p1;
#pragma unroll
            for (int off = 8; off >= 1; off >>= 1)
                rs += __shfl_xor(rs, off, 16);
            rl[g] = rl[g] * corr + rs;
            rm[g] = nm;
            s[0][g] = p0; s[1][g] = p1;
#pragma unroll
            for (int t = 0; t < 4; ++t) acc[t][g] *= corr;
        }

        // ---- P (C-layout) -> LDS -> A-layout fragment ----
        __syncthreads();
#pragma unroll
        for (int g = 0; g < 8; ++g) {
            pbuf[(hf * 8 + g) * 32 + r]      = (_Float16)s[0][g];
            pbuf[(hf * 8 + g) * 32 + 16 + r] = (_Float16)s[1][g];
        }
        __syncthreads();
        ABfrag pa;
        pa.h[0] = *(const v8h*)(&pbuf[r * 32 + koff]);
        pa.h[1] = *(const v8h*)(&pbuf[r * 32 + koff + 16]);

        // ---- O += P @ V  (V pre-transposed: contiguous key axis) ----
#pragma unroll
        for (int t = 0; t < 4; ++t) {
            const _Float16* vrow = vbase + (size_t)(t * 16 + r) * LSEQ + j0;
            ABfrag vb;
            load_frag(vb, vrow + koff);
            acc[t] = wmma_f16(pa, vb, acc[t]);
        }
    }

    // ---- normalize + store context as f16 [B, L, D] ----
#pragma unroll
    for (int g = 0; g < 8; ++g) {
        const float inv = 1.0f / rl[g];
        const int l = q0 + hf * 8 + g;
#pragma unroll
        for (int t = 0; t < 4; ++t) {
            const int d = h * HDIM + t * 16 + r;
            ctx[((size_t)b * LSEQ + l) * DMODEL + d] = (_Float16)(acc[t][g] * inv);
        }
    }
}

// ---------------------------------------------------------------------------
// Output projection: out[m,n] = sum_k ctx[m,k] * w_out[k,n] + b_out[n]  (fp32 out)
// ---------------------------------------------------------------------------
__global__ __launch_bounds__(32)
void out_gemm(const _Float16* __restrict__ ch, const _Float16* __restrict__ wT,
              const float* __restrict__ bias, float* __restrict__ out) {
    const int lane = threadIdx.x;
    const int r    = lane & 15;
    const int hf   = lane >> 4;
    const int koff = hf * 8;
    const int m0   = blockIdx.y * 32;
    const int n0   = blockIdx.x * 64;

    v8f acc[2][4];
#pragma unroll
    for (int i = 0; i < 2; ++i)
#pragma unroll
        for (int t = 0; t < 4; ++t)
#pragma unroll
            for (int g = 0; g < 8; ++g) acc[i][t][g] = 0.0f;

    gemm_tile_32x64(ch, wT, DMODEL, m0, n0, r, koff, acc);

#pragma unroll
    for (int t = 0; t < 4; ++t) {
        const int n = n0 + t * 16 + r;
        const float bb = bias[n];
#pragma unroll
        for (int i = 0; i < 2; ++i) {
            const int mbase = m0 + i * 16 + hf * 8;
#pragma unroll
            for (int g = 0; g < 8; ++g) {
                const int m = mbase + g;
                out[(size_t)m * DMODEL + n] = acc[i][t][g] + bb;
            }
        }
    }
}

// ---------------------------------------------------------------------------
extern "C" void kernel_launch(void* const* d_in, const int* in_sizes, int n_in,
                              void* d_out, int out_size, void* d_ws, size_t ws_size,
                              hipStream_t stream) {
    (void)in_sizes; (void)n_in; (void)out_size; (void)ws_size;

    const float* x     = (const float*)d_in[0];
    const float* w_qkv = (const float*)d_in[1];
    const float* b_qkv = (const float*)d_in[2];
    const float* w_out = (const float*)d_in[3];
    const float* b_out = (const float*)d_in[4];

    char* ws = (char*)d_ws;
    const size_t MB = 1024 * 1024;
    _Float16* xh    = (_Float16*)(ws + 0 * MB);    // 8 MiB  [B*L, D]
    _Float16* wqkvT = (_Float16*)(ws + 8 * MB);    // 6 MiB  [3D, D]
    _Float16* woutT = (_Float16*)(ws + 14 * MB);   // 2 MiB  [D, D]
    _Float16* qbuf  = (_Float16*)(ws + 16 * MB);   // 8 MiB  [B,H,L,HD]
    _Float16* kbuf  = (_Float16*)(ws + 24 * MB);   // 8 MiB  [B,H,L,HD]
    _Float16* vbuf  = (_Float16*)(ws + 32 * MB);   // 8 MiB  [B,H,HD,L] (transposed)
    _Float16* cbuf  = (_Float16*)(ws + 40 * MB);   // 8 MiB  [B*L, D]

    const int nx = BATCH * LSEQ * DMODEL;
    cvt_f32_f16<<<(nx + 255) / 256, 256, 0, stream>>>(x, xh, nx);

    transpose_f32_f16<<<(DMODEL * N3D + 255) / 256, 256, 0, stream>>>(w_qkv, wqkvT, DMODEL, N3D);
    transpose_f32_f16<<<(DMODEL * DMODEL + 255) / 256, 256, 0, stream>>>(w_out, woutT, DMODEL, DMODEL);

    // QKV GEMM: M = B*L = 4096 -> 128 tiles of 32, N = 3072 -> 48 blocks of 64
    qkv_gemm<<<dim3(N3D / 64, (BATCH * LSEQ) / 32), 32, 0, stream>>>(
        xh, wqkvT, b_qkv, qbuf, kbuf, vbuf);

    // Attention: one wave per (query-tile, head, batch)
    attn_window<<<dim3(LSEQ / 16, NHEAD, BATCH), 32, 0, stream>>>(qbuf, kbuf, vbuf, cbuf);

    // Output GEMM: N = 1024 -> 16 blocks of 64
    out_gemm<<<dim3(DMODEL / 64, (BATCH * LSEQ) / 32), 32, 0, stream>>>(
        cbuf, woutT, b_out, (float*)d_out);
}
